// NNCLR_9139690406168
// MI455X (gfx1250) — matrix operand order repaired and checked by reading, not compiled, not run on hardware
//
#include <hip/hip_runtime.h>

#define BATCH  512
#define DIM    256
#define QUEUE  65536
#define NCHUNK 128          // blocks in sim kernel = queue chunks
#define QT     512          // queue rows per block (LDS tile)
#define QSTR   264          // LDS row stride in halves (pad 8 -> conflict-free)
#define PROWS  1024         // p1 rows 0..511, p2 rows 512..1023

typedef __bf16 bf16_t;
typedef bf16_t v16bf __attribute__((ext_vector_type(16)));
typedef float  v8f   __attribute__((ext_vector_type(8)));

// ---------- helpers ----------
__device__ __forceinline__ unsigned bf16_bits(float f) {
  unsigned u = __float_as_uint(f);
  return (u + 0x7FFFu + ((u >> 16) & 1u)) >> 16;   // RNE
}
__device__ __forceinline__ unsigned pack2bf(float lo, float hi) {
  return (bf16_bits(hi) << 16) | (bf16_bits(lo) & 0xFFFFu);
}
union FragCast { uint4 u[2]; v16bf v; };
__device__ __forceinline__ v16bf make_frag(uint4 a, uint4 b) {
  FragCast c; c.u[0] = a; c.u[1] = b; return c.v;
}

// A fragment (16x32 bf16, ISA 7.12.2): lanes 0-15 M=lane K={0..7,16..23},
// lanes 16-31 M=lane-16 K={8..15,24..31} (+32*k). Two 16B runs per lane.
__device__ __forceinline__ v16bf load_afrag(const unsigned short* __restrict__ p,
                                            int m, int k, int half) {
  const uint4* q = reinterpret_cast<const uint4*>(p + m * DIM + k * 32 + half * 8);
  return make_frag(q[0], q[2]);   // +16 halves = +2 uint4
}
// B fragment (32x16 bf16): lane -> N = lane&15, K = 32*k + {0..7,16..23} (+8 for lane>=16)
__device__ __forceinline__ v16bf load_bfrag(const unsigned short* __restrict__ qlds,
                                            int qrow, int k, int half) {
  const uint4* q = reinterpret_cast<const uint4*>(qlds + qrow * QSTR + k * 32 + half * 8);
  return make_frag(q[0], q[2]);
}

// ---------- K1: l2-normalize, emit fp32 + bf16 ----------
__global__ __launch_bounds__(256) void normalize_kernel(
    const float* __restrict__ in1, const float* __restrict__ in2,
    float* __restrict__ pout, unsigned short* __restrict__ pbf) {
  __shared__ float sm[8];
  int row = blockIdx.x;                       // 0..1023
  int t = threadIdx.x;                        // 0..255
  const float* src = (row < BATCH) ? (in1 + row * DIM) : (in2 + (row - BATCH) * DIM);
  float x = src[t];
  float s = x * x;
  #pragma unroll
  for (int d = 16; d >= 1; d >>= 1) s += __shfl_xor(s, d, 32);
  if ((t & 31) == 0) sm[t >> 5] = s;
  __syncthreads();
  float tot = 0.f;
  #pragma unroll
  for (int i = 0; i < 8; ++i) tot += sm[i];
  float scale = 1.0f / sqrtf(fmaxf(tot, 1e-12f));
  float y = x * scale;
  pout[row * DIM + t] = y;
  pbf[row * DIM + t] = (unsigned short)bf16_bits(y);
}

// ---------- K2: fused bf16-WMMA similarity + per-chunk argmax ----------
__global__ __launch_bounds__(256, 1) void sim_argmax_kernel(
    const float* __restrict__ fq, const unsigned short* __restrict__ pbf,
    int* __restrict__ pidx) {
  __shared__ unsigned short qlds[QT * QSTR];   // 264 KB (<= 320 KB/WGP)
  const int tid = threadIdx.x;
  const int chunk = blockIdx.x;
  const int qbase = chunk * QT;

  // stage queue tile fp32 -> bf16 LDS (row-major, stride QSTR halves)
  const float4* fq4 = reinterpret_cast<const float4*>(fq);
  for (int i = tid; i < QT * (DIM / 4); i += 256) {
    int row = i >> 6, c4 = i & 63;
    float4 v = fq4[(size_t)(qbase + row) * (DIM / 4) + c4];
    uint2 pk;
    pk.x = pack2bf(v.x, v.y);
    pk.y = pack2bf(v.z, v.w);
    *reinterpret_cast<uint2*>(&qlds[row * QSTR + c4 * 4]) = pk;
  }
  __syncthreads();

  const int wave = tid >> 5;
  const int lane = tid & 31;
  const int half = lane >> 4;
  const int n16  = lane & 15;

  // each wave owns 8 M-tiles (rows wave*128 .. wave*128+127), processed in pairs
  for (int pr = 0; pr < 4; ++pr) {
    const int mt0 = wave * 8 + pr * 2;
    v16bf a0[8], a1[8];
    #pragma unroll
    for (int k = 0; k < 8; ++k) {
      a0[k] = load_afrag(pbf, (mt0    ) * 16 + n16, k, half);
      a1[k] = load_afrag(pbf, (mt0 + 1) * 16 + n16, k, half);
    }
    float best0v[8], best1v[8];
    int   best0i[8], best1i[8];
    #pragma unroll
    for (int r = 0; r < 8; ++r) {
      best0v[r] = -3.0e38f; best1v[r] = -3.0e38f;
      best0i[r] = qbase;    best1i[r] = qbase;
    }
    for (int nt = 0; nt < QT / 16; ++nt) {
      v8f acc0 = {};
      v8f acc1 = {};
      #pragma unroll
      for (int k = 0; k < 8; ++k) {
        v16bf b = load_bfrag(qlds, nt * 16 + n16, k, half);
        acc0 = __builtin_amdgcn_wmma_f32_16x16x32_bf16(false, a0[k], false, b,
                                                       (short)0, acc0, false, false);
        acc1 = __builtin_amdgcn_wmma_f32_16x16x32_bf16(false, a1[k], false, b,
                                                       (short)0, acc1, false, false);
      }
      int qidx = qbase + nt * 16 + n16;
      #pragma unroll
      for (int r = 0; r < 8; ++r) {
        if (acc0[r] > best0v[r]) { best0v[r] = acc0[r]; best0i[r] = qidx; }
        if (acc1[r] > best1v[r]) { best1v[r] = acc1[r]; best1i[r] = qidx; }
      }
    }
    // butterfly argmax across the 16-lane half (xor masks 1..8 stay in-half)
    #pragma unroll
    for (int r = 0; r < 8; ++r) {
      float v0 = best0v[r], v1 = best1v[r];
      int   i0 = best0i[r], i1 = best1i[r];
      #pragma unroll
      for (int d = 1; d < 16; d <<= 1) {
        float ov = __shfl_xor(v0, d, 32); int oi = __shfl_xor(i0, d, 32);
        if (ov > v0 || (ov == v0 && oi < i0)) { v0 = ov; i0 = oi; }
        ov = __shfl_xor(v1, d, 32); oi = __shfl_xor(i1, d, 32);
        if (ov > v1 || (ov == v1 && oi < i1)) { v1 = ov; i1 = oi; }
      }
      if (n16 == 0) {
        int row0 = mt0 * 16 + half * 8 + r;   // C-layout: VGPR r -> M=r (+8 for hi half)
        pidx[chunk * PROWS + row0]      = i0;
        pidx[chunk * PROWS + row0 + 16] = i1;
      }
    }
  }
}

// ---------- K3: fp32 rescore of 128 chunk winners + straight-through gather ----------
__global__ __launch_bounds__(128) void rescore_gather_kernel(
    const int* __restrict__ pidx, const float* __restrict__ p,
    const float* __restrict__ fq, float* __restrict__ nn) {
  __shared__ float sv[128];
  __shared__ int   si[128];
  int row = blockIdx.x;          // 0..1023
  int t = threadIdx.x;           // 0..127 (one chunk winner each)
  int cand = pidx[t * PROWS + row];
  const float4* p4 = reinterpret_cast<const float4*>(p + row * DIM);
  const float4* q4 = reinterpret_cast<const float4*>(fq + (size_t)cand * DIM);
  float s = 0.f;
  #pragma unroll 8
  for (int c = 0; c < DIM / 4; ++c) {
    float4 a = p4[c], b = q4[c];
    s += a.x * b.x + a.y * b.y + a.z * b.z + a.w * b.w;
  }
  sv[t] = s; si[t] = cand;
  __syncthreads();
  for (int off = 64; off > 0; off >>= 1) {
    if (t < off) {
      float ov = sv[t + off]; int oi = si[t + off];
      if (ov > sv[t] || (ov == sv[t] && oi < si[t])) { sv[t] = ov; si[t] = oi; }
    }
    __syncthreads();
  }
  int bi = si[0];
  const float* qr = fq + (size_t)bi * DIM;
  const float* pr = p + row * DIM;
  for (int c = t; c < DIM; c += 128) {
    float pv = pr[c];
    nn[row * DIM + c] = pv + (qr[c] - pv);   // matches p + (nn - p) rounding
  }
}

// ---------- K4: fp32 S = A * B^T / 0.1 (512x512x256), LDS-tiled ----------
__global__ __launch_bounds__(256) void matmul_nt_kernel(
    const float* __restrict__ A, const float* __restrict__ Bm,
    float* __restrict__ S) {
  __shared__ float As[16][17], Bs[16][17];
  int tx = threadIdx.x & 15, ty = threadIdx.x >> 4;
  int bi = blockIdx.x & 31, bj = blockIdx.x >> 5;   // 32x32 tiles
  int rowA = bi * 16 + ty, rowB = bj * 16 + ty;
  float acc = 0.f;
  for (int k0 = 0; k0 < DIM; k0 += 16) {
    As[ty][tx] = A[rowA * DIM + k0 + tx];
    Bs[ty][tx] = Bm[rowB * DIM + k0 + tx];
    __syncthreads();
    #pragma unroll
    for (int k = 0; k < 16; ++k) acc += As[ty][k] * Bs[tx][k];
    __syncthreads();
  }
  S[(bi * 16 + ty) * BATCH + bj * 16 + tx] = acc / 0.1f;
}

// ---------- K5: loss = LSE(row or col) - diag ----------
__global__ __launch_bounds__(256) void loss_kernel(
    const float* __restrict__ S, float* __restrict__ outRow,
    float* __restrict__ outCol) {
  __shared__ float sm[8];
  int id = blockIdx.x, t = threadIdx.x;        // 0..511 rows, 512..1023 cols
  bool isRow = id < BATCH;
  int rc = isRow ? id : id - BATCH;
  float x0, x1;
  if (isRow) { x0 = S[rc * BATCH + t];  x1 = S[rc * BATCH + t + 256]; }
  else       { x0 = S[t * BATCH + rc];  x1 = S[(t + 256) * BATCH + rc]; }
  float m = fmaxf(x0, x1);
  #pragma unroll
  for (int d = 16; d >= 1; d >>= 1) m = fmaxf(m, __shfl_xor(m, d, 32));
  if ((t & 31) == 0) sm[t >> 5] = m;
  __syncthreads();
  float mm = sm[0];
  #pragma unroll
  for (int i = 1; i < 8; ++i) mm = fmaxf(mm, sm[i]);
  __syncthreads();
  float e = expf(x0 - mm) + expf(x1 - mm);
  #pragma unroll
  for (int d = 16; d >= 1; d >>= 1) e += __shfl_xor(e, d, 32);
  if ((t & 31) == 0) sm[t >> 5] = e;
  __syncthreads();
  float ss = 0.f;
  #pragma unroll
  for (int i = 0; i < 8; ++i) ss += sm[i];
  if (t == 0) {
    float lse = mm + logf(ss);
    (isRow ? outRow : outCol)[rc] = lse - S[rc * BATCH + rc];
  }
}

// ---------- K6: new_queue = [p1 ; queue[:-512]] ----------
__global__ __launch_bounds__(256) void queue_shift_kernel(
    const float* __restrict__ p, const float* __restrict__ fq,
    float* __restrict__ outq) {
  int idx = blockIdx.x * 256 + threadIdx.x;    // float4 index, 65536*64 total
  int row = idx >> 6;
  float4 v;
  if (row < BATCH) v = reinterpret_cast<const float4*>(p)[idx];
  else             v = reinterpret_cast<const float4*>(fq)[idx - BATCH * (DIM / 4)];
  reinterpret_cast<float4*>(outq)[idx] = v;
}

extern "C" void kernel_launch(void* const* d_in, const int* in_sizes, int n_in,
                              void* d_out, int out_size, void* d_ws, size_t ws_size,
                              hipStream_t stream) {
  const float* proj1 = (const float*)d_in[0];
  const float* proj2 = (const float*)d_in[1];
  const float* fq    = (const float*)d_in[2];
  float* loss = (float*)d_out;                 // [2048]
  float* outq = (float*)d_out + 4 * BATCH;     // [65536*256]

  char* ws = (char*)d_ws;                      // ~5 MB used
  float*          p_f32 = (float*)(ws);                                  // 1 MB
  unsigned short* p_bf  = (unsigned short*)(ws + (1u << 20));            // 512 KB
  int*            pidx  = (int*)(ws + (1u << 20) + (512u << 10));        // 512 KB
  float*          nn    = (float*)(ws + (2u << 20));                     // 1 MB
  float*          S1    = (float*)(ws + (3u << 20));                     // 1 MB
  float*          S2    = (float*)(ws + (4u << 20));                     // 1 MB

  normalize_kernel<<<PROWS, 256, 0, stream>>>(proj1, proj2, p_f32, p_bf);
  sim_argmax_kernel<<<NCHUNK, 256, 0, stream>>>(fq, p_bf, pidx);
  rescore_gather_kernel<<<PROWS, 128, 0, stream>>>(pidx, p_f32, fq, nn);
  matmul_nt_kernel<<<1024, 256, 0, stream>>>(nn, p_f32 + BATCH * DIM, S1);          // nn1 . p2^T
  matmul_nt_kernel<<<1024, 256, 0, stream>>>(nn + BATCH * DIM, p_f32, S2);          // nn2 . p1^T
  loss_kernel<<<1024, 256, 0, stream>>>(S1, loss, loss + BATCH);
  loss_kernel<<<1024, 256, 0, stream>>>(S2, loss + 2 * BATCH, loss + 3 * BATCH);
  queue_shift_kernel<<<QUEUE * (DIM / 4) / 256, 256, 0, stream>>>(p_f32, fq, outq);
}
